// MultiqueryAttention_59940563583325
// MI455X (gfx1250) — matrix operand order repaired and checked by reading, not compiled, not run on hardware
//
#include <hip/hip_runtime.h>
#include <hip/hip_bf16.h>
#include <math.h>

// ---------------------------------------------------------------------------
// MultiqueryAttention for MI455X (gfx1250, wave32, WMMA).
// bf16 matmuls w/ f32 accumulation via v_wmma_f32_16x16x32_bf16.
// Weights and V pre-swizzled into the WMMA B-fragment register layout so the
// hot loops are pure global_load_b128 -> v_wmma streams.
// Flash attention: 64-wide key steps, exp2-domain softmax (scale folded into
// the Q projection), and DPP-based 16-lane reductions (no LDS round-trips).
// ---------------------------------------------------------------------------

typedef __bf16 bf16_t;
typedef __bf16 v16bf __attribute__((ext_vector_type(16)));
typedef float  v8f   __attribute__((ext_vector_type(8)));

#define EMBD_C  1024
#define HEADS_C 16
#define HD_C    64
#define BATCH_C 4
#define SEQ_C   2048

union FragBF { v16bf v; uint4 u[2]; };

__device__ __forceinline__ v8f v8f_zero() {
  v8f z;
#pragma unroll
  for (int i = 0; i < 8; ++i) z[i] = 0.0f;
  return z;
}

// ---- DPP 16-lane butterfly reductions (VALU only, rows of 16 stay separate)
template <int CTRL>
__device__ __forceinline__ float dpp_movf(float x) {
  return __int_as_float(__builtin_amdgcn_update_dpp(
      0, __float_as_int(x), CTRL, 0xF, 0xF, true));
}
__device__ __forceinline__ float red_max16(float x) {
  x = fmaxf(x, dpp_movf<0xB1>(x));    // quad_perm [1,0,3,2]  (xor 1)
  x = fmaxf(x, dpp_movf<0x4E>(x));    // quad_perm [2,3,0,1]  (xor 2)
  x = fmaxf(x, dpp_movf<0x141>(x));   // row_half_mirror      (cross quads in 8)
  x = fmaxf(x, dpp_movf<0x140>(x));   // row_mirror           (cross 8s in 16)
  return x;
}
__device__ __forceinline__ float red_add16(float x) {
  x += dpp_movf<0xB1>(x);
  x += dpp_movf<0x4E>(x);
  x += dpp_movf<0x141>(x);
  x += dpp_movf<0x140>(x);
  return x;
}

// ---------------------------------------------------------------------------
// fp32 -> bf16 cast (4 elements / thread; sizes are multiples of 4)
// ---------------------------------------------------------------------------
__global__ __launch_bounds__(256) void cast_f32_to_bf16_k(
    const float* __restrict__ s, bf16_t* __restrict__ d, int n) {
  int i = (blockIdx.x * 256 + threadIdx.x) * 4;
  if (i + 3 < n) {
    float4 f = *reinterpret_cast<const float4*>(s + i);
    d[i + 0] = (bf16_t)f.x;
    d[i + 1] = (bf16_t)f.y;
    d[i + 2] = (bf16_t)f.z;
    d[i + 3] = (bf16_t)f.w;
  } else {
    for (; i < n; ++i) d[i] = (bf16_t)s[i];
  }
}

// ---------------------------------------------------------------------------
// Weight swizzle: W[K][N] fp32 row-major -> bf16 WMMA B-fragment layout.
// Fragment f = (k/32)*(N/16) + (n/16).  Within fragment: 32 lanes x 16 elems
// contiguous.  lane = 16*((k>>4)&1) + (n&15), elem e = k&15.
// ---------------------------------------------------------------------------
__global__ __launch_bounds__(256) void swizzle_w_k(
    const float* __restrict__ W, bf16_t* __restrict__ Wsw, int N, int total) {
  int tid = blockIdx.x * 256 + threadIdx.x;   // (frag, lane) pairs
  if (tid >= total) return;
  const int lane   = tid & 31;
  const int f      = tid >> 5;
  const int nt16   = N >> 4;
  const int ntile  = f % nt16;
  const int kb5    = f / nt16;
  const int hsel   = lane >> 4;
  const int l15    = lane & 15;
  const float* src = W + (size_t)(kb5 * 32 + hsel * 16) * N + ntile * 16 + l15;
  bf16_t* dst = Wsw + (size_t)tid * 16;
#pragma unroll
  for (int e = 0; e < 16; ++e) dst[e] = (bf16_t)src[(size_t)e * N];
}

// ---------------------------------------------------------------------------
// V swizzle: V[b*SEQ + t][64] bf16 -> B-fragment layout per batch:
// frag = ((b*(SEQ/32) + t/32)*4 + d/16); lane = 16*((t>>4)&1) + (d&15); e = t&15
// ---------------------------------------------------------------------------
__global__ __launch_bounds__(256) void swizzle_v_k(
    const bf16_t* __restrict__ V, bf16_t* __restrict__ Vsw) {
  int tid = blockIdx.x * 256 + threadIdx.x;   // BATCH*(SEQ/32)*4*32 = 32768
  const int lane  = tid & 31;
  const int f     = tid >> 5;
  const int dtile = f & 3;
  const int bt    = f >> 2;
  const int t5    = bt & (SEQ_C / 32 - 1);
  const int b     = bt >> 6;                  // SEQ/32 = 64
  const int hsel  = lane >> 4;
  const int l15   = lane & 15;
  const bf16_t* src =
      V + ((size_t)b * SEQ_C + t5 * 32 + hsel * 16) * HD_C + dtile * 16 + l15;
  bf16_t* dst = Vsw + (size_t)tid * 16;
#pragma unroll
  for (int e = 0; e < 16; ++e) dst[e] = src[(size_t)e * HD_C];
}

// ---------------------------------------------------------------------------
// bf16 GEMM: C[M,N] = ((A[M,K] * Bsw) + bias) * oscale.  Bsw frag-swizzled.
// One wave -> 32x64 tile: 2 A-frags x 4 B-frags -> 8 WMMAs / k-step.
// ---------------------------------------------------------------------------
template <bool OUT_F32>
__global__ __launch_bounds__(256) void gemm_bf16_k(
    const bf16_t* __restrict__ A, const bf16_t* __restrict__ Bsw,
    const float* __restrict__ bias, void* __restrict__ outp,
    int M, int N, int K, float oscale) {
  const int lane = threadIdx.x & 31;
  const int wave = blockIdx.x * 8 + (threadIdx.x >> 5);
  const int tiles_n = N >> 6;                 // 64-wide wave tiles
  const int tm = wave / tiles_n;
  const int tn = wave % tiles_n;
  if (tm * 32 >= M) return;                   // wave-uniform, EXEC stays full
  const int hsel = lane >> 4;
  const int l15  = lane & 15;
  const int nt16 = N >> 4;

  v8f acc[2][4];
#pragma unroll
  for (int i = 0; i < 2; ++i)
#pragma unroll
    for (int j = 0; j < 4; ++j) acc[i][j] = v8f_zero();

  const bf16_t* Arow0 = A + (size_t)(tm * 32 + l15) * K;
  const bf16_t* Arow1 = Arow0 + (size_t)16 * K;

  for (int kb5 = 0; kb5 < (K >> 5); ++kb5) {
    const int kb = kb5 << 5;
    FragBF a0, a1;
    const bf16_t* ap0 = Arow0 + kb + hsel * 16;
    a0.u[0] = *reinterpret_cast<const uint4*>(ap0);
    a0.u[1] = *reinterpret_cast<const uint4*>(ap0 + 8);
    const bf16_t* ap1 = Arow1 + kb + hsel * 16;
    a1.u[0] = *reinterpret_cast<const uint4*>(ap1);
    a1.u[1] = *reinterpret_cast<const uint4*>(ap1 + 8);
    const bf16_t* brow =
        Bsw + ((size_t)(kb5 * nt16 + tn * 4) * 32 + lane) * 16;
#pragma unroll
    for (int j = 0; j < 4; ++j) {
      FragBF bfrag;
      const bf16_t* bp = brow + (size_t)j * 512;   // 32 lanes * 16 elems
      bfrag.u[0] = *reinterpret_cast<const uint4*>(bp);
      bfrag.u[1] = *reinterpret_cast<const uint4*>(bp + 8);
      acc[0][j] = __builtin_amdgcn_wmma_f32_16x16x32_bf16(
          false, a0.v, false, bfrag.v, (short)0, acc[0][j], false, false);
      acc[1][j] = __builtin_amdgcn_wmma_f32_16x16x32_bf16(
          false, a1.v, false, bfrag.v, (short)0, acc[1][j], false, false);
    }
  }

#pragma unroll
  for (int i = 0; i < 2; ++i) {
#pragma unroll
    for (int j = 0; j < 4; ++j) {
      const int n = tn * 64 + j * 16 + l15;
      const float bv = bias ? bias[n] : 0.0f;
#pragma unroll
      for (int e = 0; e < 8; ++e) {
        const int row = tm * 32 + i * 16 + e + hsel * 8;
        const float v = (acc[i][j][e] + bv) * oscale;
        if (OUT_F32) ((float*)outp)[(size_t)row * N + n] = v;
        else         ((bf16_t*)outp)[(size_t)row * N + n] = (bf16_t)v;
      }
    }
  }
}

// ---------------------------------------------------------------------------
// Flash multi-query attention.  One wave per (b, h, 16-row query block).
// Keys streamed 64 at a time (4 score tiles / step).  Q pre-scaled by
// log2(e)/sqrt(Dh) -> softmax on bare v_exp_f32.  Row reductions via DPP
// butterflies (VALU, co-executes with WMMA).  P transposed D->A layout via
// per-wave LDS (pitch 72 bf16, conflict-free b128 reads); V from swizzled
// fragment layout.
// ---------------------------------------------------------------------------
__global__ __launch_bounds__(256) void mqa_flash_k(
    const bf16_t* __restrict__ Q, const bf16_t* __restrict__ Km,
    const bf16_t* __restrict__ Vsw, bf16_t* __restrict__ O) {
  __shared__ bf16_t Plds[8 * 16 * 72];        // 18432 B, 2304 B / wave

  const int lane  = threadIdx.x & 31;
  const int wslot = threadIdx.x >> 5;
  const int wave  = blockIdx.x * 8 + wslot;
  const int SB    = SEQ_C / 16;               // 128 query blocks
  const int sblk  = wave % SB;
  const int bh    = wave / SB;
  const int h     = bh % HEADS_C;
  const int b     = bh / HEADS_C;
  const int hsel  = lane >> 4;
  const int l15   = lane & 15;

  // ---- Q A-fragments (two K=32 chunks covering d = 0..63) ----
  FragBF qa[2];
  {
    const int s = sblk * 16 + l15;
    const bf16_t* qp = Q + (size_t)(b * SEQ_C + s) * EMBD_C + h * HD_C;
#pragma unroll
    for (int kc = 0; kc < 2; ++kc) {
      const bf16_t* p = qp + kc * 32 + hsel * 8;
      qa[kc].u[0] = *reinterpret_cast<const uint4*>(p);
      qa[kc].u[1] = *reinterpret_cast<const uint4*>(p + 16);
    }
  }

  float mrow[8], lrow[8];
  v8f acc[4];
#pragma unroll
  for (int e = 0; e < 8; ++e) { mrow[e] = -3.402823466e38f; lrow[e] = 0.0f; }
#pragma unroll
  for (int j = 0; j < 4; ++j) acc[j] = v8f_zero();

  bf16_t* pl = Plds + wslot * 16 * 72;
  const bf16_t* Kbase  = Km + (size_t)b * SEQ_C * HD_C;
  const bf16_t* Vbatch = Vsw + (size_t)b * (SEQ_C / 32) * 4 * 512;

  for (int t0 = 0; t0 < SEQ_C; t0 += 64) {
    // ---- scores (log2 domain): four 16x16 tiles covering t0..t0+63 ----
    v8f st[4];
#pragma unroll
    for (int tl = 0; tl < 4; ++tl) {
      v8f c = v8f_zero();
      const int t = t0 + tl * 16 + l15;
      const bf16_t* kp = Kbase + (size_t)t * HD_C;
#pragma unroll
      for (int kc = 0; kc < 2; ++kc) {
        FragBF kf;
        const bf16_t* p = kp + kc * 32 + hsel * 8;
        kf.u[0] = *reinterpret_cast<const uint4*>(p);
        kf.u[1] = *reinterpret_cast<const uint4*>(p + 16);
        c = __builtin_amdgcn_wmma_f32_16x16x32_bf16(
            false, qa[kc].v, false, kf.v, (short)0, c, false, false);
      }
      st[tl] = c;
    }

    // ---- online softmax in exp2 domain (DPP reductions, rows in halves) ---
    float corr[8];
#pragma unroll
    for (int e = 0; e < 8; ++e) {
      float tmax = fmaxf(fmaxf(st[0][e], st[1][e]), fmaxf(st[2][e], st[3][e]));
      tmax = red_max16(tmax);
      const float mn = fmaxf(mrow[e], tmax);
      corr[e] = exp2f(mrow[e] - mn);
      mrow[e] = mn;
      float rs = 0.0f;
#pragma unroll
      for (int tl = 0; tl < 4; ++tl) {
        const float p = exp2f(st[tl][e] - mn);
        st[tl][e] = p;
        rs += p;
      }
      rs = red_add16(rs);
      lrow[e] = lrow[e] * corr[e] + rs;
    }
#pragma unroll
    for (int j = 0; j < 4; ++j)
#pragma unroll
      for (int e = 0; e < 8; ++e) acc[j][e] *= corr[e];

    // ---- P: D-layout -> LDS -> A-layout (same-wave DS ops are in-order) ----
#pragma unroll
    for (int tl = 0; tl < 4; ++tl)
#pragma unroll
      for (int e = 0; e < 8; ++e)
        pl[(e + hsel * 8) * 72 + tl * 16 + l15] = (bf16_t)st[tl][e];
    __builtin_amdgcn_wave_barrier();
    asm volatile("" ::: "memory");
    FragBF pf[2];
#pragma unroll
    for (int kc = 0; kc < 2; ++kc) {
      const bf16_t* pr = pl + l15 * 72 + kc * 32 + hsel * 16;
      pf[kc].u[0] = *reinterpret_cast<const uint4*>(pr);
      pf[kc].u[1] = *reinterpret_cast<const uint4*>(pr + 8);
    }

    // ---- accumulate: out += P (16x64) * V (64x64), V from swizzled layout --
#pragma unroll
    for (int kc = 0; kc < 2; ++kc) {
      const bf16_t* vrow =
          Vbatch + (((size_t)((t0 >> 5) + kc) * 4) * 32 + lane) * 16;
#pragma unroll
      for (int j = 0; j < 4; ++j) {
        FragBF vf;
        const bf16_t* vp = vrow + (size_t)j * 512;
        vf.u[0] = *reinterpret_cast<const uint4*>(vp);
        vf.u[1] = *reinterpret_cast<const uint4*>(vp + 8);
        acc[j] = __builtin_amdgcn_wmma_f32_16x16x32_bf16(
            false, pf[kc].v, false, vf.v, (short)0, acc[j], false, false);
      }
    }
  }

  // ---- normalize and store bf16 [B,S,EMBD] (head-interleaved) ----
#pragma unroll
  for (int j = 0; j < 4; ++j) {
#pragma unroll
    for (int e = 0; e < 8; ++e) {
      const int s = sblk * 16 + e + hsel * 8;
      const float v = acc[j][e] / lrow[e];
      O[(size_t)(b * SEQ_C + s) * EMBD_C + h * HD_C + j * 16 + l15] = (bf16_t)v;
    }
  }
}

// ---------------------------------------------------------------------------
// Launcher
// ---------------------------------------------------------------------------
extern "C" void kernel_launch(void* const* d_in, const int* in_sizes, int n_in,
                              void* d_out, int out_size, void* d_ws, size_t ws_size,
                              hipStream_t stream) {
  const float* x  = (const float*)d_in[0];
  const float* Wq = (const float*)d_in[1];
  const float* bq = (const float*)d_in[2];
  const float* Wk = (const float*)d_in[3];
  const float* bk = (const float*)d_in[4];
  const float* Wv = (const float*)d_in[5];
  const float* bv = (const float*)d_in[6];
  const float* Wo = (const float*)d_in[7];
  const float* bo = (const float*)d_in[8];
  float* out = (float*)d_out;

  const int M    = BATCH_C * SEQ_C;           // 8192 rows
  const int n_x  = M * EMBD_C;                // 8,388,608
  const int n_Wq = EMBD_C * EMBD_C;
  const int n_Wk = EMBD_C * HD_C;

  char* ws = (char*)d_ws;
  size_t off = 0;
  auto bump = [&](size_t bytes) -> char* {
    char* p = ws + off;
    off = (off + bytes + 255) & ~(size_t)255;
    return p;
  };
  bf16_t* xb  = (bf16_t*)bump((size_t)n_x  * 2);
  bf16_t* Wqb = (bf16_t*)bump((size_t)n_Wq * 2);   // swizzled
  bf16_t* Wkb = (bf16_t*)bump((size_t)n_Wk * 2);   // swizzled
  bf16_t* Wvb = (bf16_t*)bump((size_t)n_Wk * 2);   // swizzled
  bf16_t* Wob = (bf16_t*)bump((size_t)n_Wq * 2);   // swizzled
  bf16_t* Qb  = (bf16_t*)bump((size_t)M * EMBD_C * 2);
  bf16_t* Kb  = (bf16_t*)bump((size_t)M * HD_C * 2);
  bf16_t* Vb  = (bf16_t*)bump((size_t)M * HD_C * 2);
  bf16_t* Vsw = (bf16_t*)bump((size_t)M * HD_C * 2);
  bf16_t* Ob  = (bf16_t*)bump((size_t)M * EMBD_C * 2);

  // ---- cast activations; swizzle weights into B-fragment layout ----
  cast_f32_to_bf16_k<<<(n_x / 4 + 255) / 256, 256, 0, stream>>>(x, xb, n_x);
  {
    const int tq = (EMBD_C / 32) * (EMBD_C / 16) * 32;   // 65536 (frag,lane)
    const int tk = (EMBD_C / 32) * (HD_C / 16) * 32;     // 4096
    swizzle_w_k<<<(tq + 255) / 256, 256, 0, stream>>>(Wq, Wqb, EMBD_C, tq);
    swizzle_w_k<<<(tk + 255) / 256, 256, 0, stream>>>(Wk, Wkb, HD_C,  tk);
    swizzle_w_k<<<(tk + 255) / 256, 256, 0, stream>>>(Wv, Wvb, HD_C,  tk);
    swizzle_w_k<<<(tq + 255) / 256, 256, 0, stream>>>(Wo, Wob, EMBD_C, tq);
  }

  // ---- Q/K/V projections (32x64 tile / wave, 8 waves / block) ----
  // Q carries log2(e)/sqrt(Dh) so attention softmax runs in the exp2 domain.
  const float qscale = 1.44269504088896340736f * 0.125f;
  {
    int waves_q = (M / 32) * (EMBD_C / 64);   // 4096
    gemm_bf16_k<false><<<waves_q / 8, 256, 0, stream>>>(
        xb, Wqb, bq, (void*)Qb, M, EMBD_C, EMBD_C, qscale);
    int waves_kv = (M / 32) * (HD_C / 64);    // 256
    gemm_bf16_k<false><<<waves_kv / 8, 256, 0, stream>>>(
        xb, Wkb, bk, (void*)Kb, M, HD_C, EMBD_C, 1.0f);
    gemm_bf16_k<false><<<waves_kv / 8, 256, 0, stream>>>(
        xb, Wvb, bv, (void*)Vb, M, HD_C, EMBD_C, 1.0f);
  }

  // ---- V -> B-fragment layout ----
  {
    const int tv = BATCH_C * (SEQ_C / 32) * 4 * 32;      // 32768
    swizzle_v_k<<<(tv + 255) / 256, 256, 0, stream>>>(Vb, Vsw);
  }

  // ---- flash attention: one wave per (b, h, 16-row query block) ----
  {
    int waves = BATCH_C * HEADS_C * (SEQ_C / 16);        // 8192
    mqa_flash_k<<<waves / 8, 256, 0, stream>>>(Qb, Kb, Vsw, Ob);
  }

  // ---- output projection -> fp32 d_out ----
  {
    int waves = (M / 32) * (EMBD_C / 64);     // 4096
    gemm_bf16_k<true><<<waves / 8, 256, 0, stream>>>(
        Ob, Wob, bo, (void*)out, M, EMBD_C, EMBD_C, 1.0f);
  }
}
